// DenoiseWavenetCondition_8890582303450
// MI455X (gfx1250) — matrix operand order repaired and checked by reading, not compiled, not run on hardware
//
#include <hip/hip_runtime.h>
#include <hip/hip_bf16.h>

typedef __attribute__((ext_vector_type(16))) _Float16 v16h;
typedef __attribute__((ext_vector_type(8)))  float    v8f;
typedef __attribute__((ext_vector_type(4)))  uint32_t u32x4;

#define TLEN 2048
#define CCH  128
#define BB   8
#define LNUM 10
#define MT   (BB * TLEN)        // 16384 rows (b*T+t)
#define PADH 512                // max dilation
#define HROWS (TLEN + 2*PADH)   // 3072 padded rows per batch for h
#define SROWS (TLEN + 2)        // 2050 padded rows per batch for skip / out1

union Frag16 {
  v16h v;
  _Float16 h[16];
  uint32_t u[8];
  u32x4 q[2];
};

// A fragment (16x32 f16), per ISA layout: two contiguous 16B runs per lane.
__device__ __forceinline__ void load_a_frag(Frag16& f, const _Float16* __restrict__ rowPtr, int hf) {
  f.q[0] = *(const u32x4*)(rowPtr + hf * 8);
  f.q[1] = *(const u32x4*)(rowPtr + 16 + hf * 8);
}

// B fragment from pre-packed weights: 32 contiguous bytes per lane.
__device__ __forceinline__ void load_b_packed(Frag16& f, const _Float16* __restrict__ p) {
  f.q[0] = ((const u32x4*)p)[0];
  f.q[1] = ((const u32x4*)p)[1];
}

#define WMMA_F16(acc, a, b) \
  acc = __builtin_amdgcn_wmma_f32_16x16x32_f16(false, (a).v, false, (b).v, (short)0, acc, false, false)

// ---------------- packing / utility kernels ----------------

// Pack f32 weights [mats][K][N] -> fragment-major f16.
__global__ void k_pack_b(const float* __restrict__ src, _Float16* __restrict__ dst,
                         int K, int N, int totalThreads) {
  int gid = blockIdx.x * blockDim.x + threadIdx.x;
  if (gid >= totalThreads) return;
  int lane = gid & 31;
  int tIdx = gid >> 5;
  int ktiles = K >> 5, ntiles = N >> 4;
  int kt = tIdx % ktiles;
  int rest = tIdx / ktiles;            // mat*ntiles + nt
  int nt = rest % ntiles;
  int mat = rest / ntiles;
  const float* S = src + (size_t)mat * K * N;
  int krow = kt * 32 + (lane >> 4) * 8;
  int col = nt * 16 + (lane & 15);
  Frag16 f;
#pragma unroll
  for (int e = 0; e < 16; ++e)
    f.h[e] = (_Float16)S[(size_t)(krow + e + ((e >> 3) << 3)) * N + col];
  _Float16* D = dst + (size_t)gid * 16;
  ((u32x4*)D)[0] = f.q[0];
  ((u32x4*)D)[1] = f.q[1];
}

__global__ void k_zero_f16(_Float16* __restrict__ p, int n) {
  int i = blockIdx.x * blockDim.x + threadIdx.x;
  if (i < n) p[i] = (_Float16)0.0f;
}

// zero the +/-1 pad rows of the padded out1 buffer
__global__ void k_zero_o1_pads(_Float16* __restrict__ o1P) {
  int i = blockIdx.x * blockDim.x + threadIdx.x;   // over BB*2*2048
  int c = i & 2047;
  int r = (i >> 11) & 1;
  int b = i >> 12;
  size_t row = (size_t)b * SROWS + (r ? (SROWS - 1) : 0);
  o1P[row * 2048 + c] = (_Float16)0.0f;
}

// h = x*Wc + bc : writes f32 (m-indexed) + padded f16 ; zeroes skip accumulator
__global__ void k_input_proj(const float* __restrict__ x, const float* __restrict__ Wc,
                             const float* __restrict__ bc, float* __restrict__ hF32,
                             _Float16* __restrict__ hPad, float* __restrict__ skipF32) {
  int i = blockIdx.x * blockDim.x + threadIdx.x;   // over MT*CCH
  int c = i & (CCH - 1);
  int m = i >> 7;
  int b = m >> 11, t = m & (TLEN - 1);
  float v = x[m] * Wc[c] + bc[c];
  hF32[i] = v;
  hPad[((size_t)b * HROWS + PADH + t) * CCH + c] = (_Float16)v;
  skipF32[i] = 0.0f;
}

// skipF32 (m-indexed) -> padded f16
__global__ void k_skip_to_pad(const float* __restrict__ skipF32, _Float16* __restrict__ skPad) {
  int i = blockIdx.x * blockDim.x + threadIdx.x;   // over MT*CCH
  int c = i & (CCH - 1);
  int m = i >> 7;
  int b = m >> 11, t = m & (TLEN - 1);
  skPad[((size_t)b * SROWS + 1 + t) * CCH + c] = (_Float16)skipF32[i];
}

// tt[b,t,k] = Bt[l,t,k] + sum_c cond[b,c]*Dt[l,c,t,k]  (same for ts)
// Pure HBM stream (33.5 MB Dt/Ds per block), b128 vectorized.
__global__ void k_cond_dense(const float* __restrict__ cond,
                             const float* __restrict__ Dt, const float* __restrict__ Bt,
                             const float* __restrict__ Ds, const float* __restrict__ Bs,
                             float* __restrict__ tt, float* __restrict__ ts, int l) {
  int i = blockIdx.x * blockDim.x + threadIdx.x;   // over MT*CCH/4
  int kq = (i & 31) << 2;
  int t  = (i >> 5) & (TLEN - 1);
  int b  = i >> 16;
  float cb[16];
#pragma unroll
  for (int c = 0; c < 16; ++c) cb[c] = cond[b * 16 + c];
  size_t bofs = ((size_t)l * TLEN + t) * CCH + kq;
  float4 at = *(const float4*)(Bt + bofs);
  float4 as = *(const float4*)(Bs + bofs);
#pragma unroll
  for (int c = 0; c < 16; ++c) {
    size_t idx = (((size_t)(l * 16 + c)) * TLEN + t) * CCH + kq;
    float4 dt = *(const float4*)(Dt + idx);
    float4 ds = *(const float4*)(Ds + idx);
    float w = cb[c];
    at.x += w * dt.x; at.y += w * dt.y; at.z += w * dt.z; at.w += w * dt.w;
    as.x += w * ds.x; as.y += w * ds.y; as.z += w * ds.z; as.w += w * ds.w;
  }
  size_t o = ((size_t)(b * TLEN + t)) * CCH + kq;
  *(float4*)(tt + o) = at;
  *(float4*)(ts + o) = as;
}

// ---------------- WMMA GEMM kernels (all branch-free inner loops) ----------------

// Dilated gated conv: two M tiles + two branches per wave (4 accumulators).
__global__ void __launch_bounds__(256)
k_gated_conv(const _Float16* __restrict__ hPad, const _Float16* __restrict__ WtP,
             const _Float16* __restrict__ WsP, const float* __restrict__ bt,
             const float* __restrict__ bs, const float* __restrict__ tt,
             const float* __restrict__ ts, _Float16* __restrict__ gF16,
             int l, int dil) {
  const int lane = threadIdx.x & 31;
  const int wave = threadIdx.x >> 5;
  const int hf = lane >> 4, ln = lane & 15;
  int tile = blockIdx.x * 8 + wave;                   // over (MT/32)*8
  int tileM = tile >> 3, tileN = tile & 7;
  int mBase = tileM * 32, nBase = tileN * 16;
  int row0 = mBase + ln, row1 = row0 + 16;
  int b = row0 >> 11;                                 // both rows in same batch
  int t0 = row0 & (TLEN - 1), t1 = row1 & (TLEN - 1);

  v8f accT0 = {}, accT1 = {}, accS0 = {}, accS1 = {};
  for (int j = 0; j < 3; ++j) {
    int d = (j - 1) * dil;
    const _Float16* aRow0 = hPad + ((size_t)b * HROWS + PADH + t0 + d) * CCH;
    const _Float16* aRow1 = hPad + ((size_t)b * HROWS + PADH + t1 + d) * CCH;
    int mat = l * 3 + j;
    const _Float16* bTp = WtP + ((size_t)(mat * 8 + tileN) * 4) * 512 + lane * 16;
    const _Float16* bSp = WsP + ((size_t)(mat * 8 + tileN) * 4) * 512 + lane * 16;
#pragma unroll
    for (int kt = 0; kt < 4; ++kt) {
      Frag16 a0, a1, bT, bS;
      load_a_frag(a0, aRow0 + kt * 32, hf);
      load_a_frag(a1, aRow1 + kt * 32, hf);
      load_b_packed(bT, bTp + kt * 512);
      load_b_packed(bS, bSp + kt * 512);
      WMMA_F16(accT0, a0, bT);
      WMMA_F16(accT1, a1, bT);
      WMMA_F16(accS0, a0, bS);
      WMMA_F16(accS1, a1, bS);
    }
  }
  int c = nBase + ln;
  float btc = bt[l * CCH + c];
  float bsc = bs[l * CCH + c];
#pragma unroll
  for (int e = 0; e < 8; ++e) {                       // D: M = e + 8*hf
    int m0 = mBase + hf * 8 + e;
    int m1 = m0 + 16;
    size_t o0 = (size_t)m0 * CCH + c;
    size_t o1 = (size_t)m1 * CCH + c;
    float vt0 = accT0[e] + btc + tt[o0];
    float vs0 = accS0[e] + bsc + ts[o0];
    float vt1 = accT1[e] + btc + tt[o1];
    float vs1 = accS1[e] + bsc + ts[o1];
    gF16[o0] = (_Float16)(tanhf(vt0) * (1.0f / (1.0f + __expf(-vs0))));
    gF16[o1] = (_Float16)(tanhf(vt1) * (1.0f / (1.0f + __expf(-vs1))));
  }
}

// skip += g@Wskip + bskip ; h = g@Wres + bres + h  (two M tiles per wave)
__global__ void __launch_bounds__(256)
k_skip_res(const _Float16* __restrict__ gF16, const _Float16* __restrict__ WskP,
           const _Float16* __restrict__ WreP, const float* __restrict__ bskip,
           const float* __restrict__ bres, float* __restrict__ skipF32,
           float* __restrict__ hF32, _Float16* __restrict__ hPad, int l) {
  const int lane = threadIdx.x & 31;
  const int wave = threadIdx.x >> 5;
  const int hf = lane >> 4, ln = lane & 15;
  int tile = blockIdx.x * 8 + wave;                   // over (MT/32)*8
  int tileM = tile >> 3, tileN = tile & 7;
  int mBase = tileM * 32, nBase = tileN * 16;
  int row0 = mBase + ln, row1 = row0 + 16;

  v8f accK0 = {}, accK1 = {}, accR0 = {}, accR1 = {};
  const _Float16* aRow0 = gF16 + (size_t)row0 * CCH;
  const _Float16* aRow1 = gF16 + (size_t)row1 * CCH;
  const _Float16* bKp = WskP + ((size_t)(l * 8 + tileN) * 4) * 512 + lane * 16;
  const _Float16* bRp = WreP + ((size_t)(l * 8 + tileN) * 4) * 512 + lane * 16;
#pragma unroll
  for (int kt = 0; kt < 4; ++kt) {
    Frag16 a0, a1, bK, bR;
    load_a_frag(a0, aRow0 + kt * 32, hf);
    load_a_frag(a1, aRow1 + kt * 32, hf);
    load_b_packed(bK, bKp + kt * 512);
    load_b_packed(bR, bRp + kt * 512);
    WMMA_F16(accK0, a0, bK);
    WMMA_F16(accK1, a1, bK);
    WMMA_F16(accR0, a0, bR);
    WMMA_F16(accR1, a1, bR);
  }
  int c = nBase + ln;
  float bkc = bskip[l * CCH + c];
  float brc = bres[l * CCH + c];
#pragma unroll
  for (int e = 0; e < 8; ++e) {
    int m0 = mBase + hf * 8 + e;
    int m1 = m0 + 16;
    int b0 = m0 >> 11, t0 = m0 & (TLEN - 1);
    int b1 = m1 >> 11, t1 = m1 & (TLEN - 1);
    size_t o0 = (size_t)m0 * CCH + c;
    size_t o1 = (size_t)m1 * CCH + c;
    skipF32[o0] += accK0[e] + bkc;
    skipF32[o1] += accK1[e] + bkc;
    float hn0 = accR0[e] + brc + hF32[o0];
    float hn1 = accR1[e] + brc + hF32[o1];
    hF32[o0] = hn0;
    hF32[o1] = hn1;
    hPad[((size_t)b0 * HROWS + PADH + t0) * CCH + c] = (_Float16)hn0;
    hPad[((size_t)b1 * HROWS + PADH + t1) * CCH + c] = (_Float16)hn1;
  }
}

// head1: out1 = relu(conv_k3(skip, W1) + b1)  128 -> 2048, two M tiles per wave
__global__ void __launch_bounds__(256)
k_head1(const _Float16* __restrict__ skPad, const _Float16* __restrict__ W1P,
        const float* __restrict__ bias1, _Float16* __restrict__ o1Pad) {
  const int lane = threadIdx.x & 31;
  const int wave = threadIdx.x >> 5;
  const int hf = lane >> 4, ln = lane & 15;
  const int nTilesN = 2048 / 16;                      // 128
  int tile = blockIdx.x * 8 + wave;                   // over (MT/32)*128
  int tileM = tile / nTilesN, tileN = tile % nTilesN;
  int mBase = tileM * 32, nBase = tileN * 16;
  int row0 = mBase + ln, row1 = row0 + 16;
  int b = row0 >> 11;
  int t0 = row0 & (TLEN - 1), t1 = row1 & (TLEN - 1);

  v8f acc0 = {}; v8f acc1 = {};
  for (int j = 0; j < 3; ++j) {
    const _Float16* aRow0 = skPad + ((size_t)b * SROWS + t0 + j) * CCH;  // 1+t0+(j-1)
    const _Float16* aRow1 = skPad + ((size_t)b * SROWS + t1 + j) * CCH;
    const _Float16* bp = W1P + ((size_t)(j * 128 + tileN) * 4) * 512 + lane * 16;
#pragma unroll
    for (int kt = 0; kt < 4; ++kt) {
      Frag16 a0, a1, bf;
      load_a_frag(a0, aRow0 + kt * 32, hf);
      load_a_frag(a1, aRow1 + kt * 32, hf);
      load_b_packed(bf, bp + kt * 512);
      WMMA_F16(acc0, a0, bf);
      WMMA_F16(acc1, a1, bf);
    }
  }
  int c = nBase + ln;
  float bc1 = bias1[c];
#pragma unroll
  for (int e = 0; e < 8; ++e) {
    int m0 = mBase + hf * 8 + e;
    int m1 = m0 + 16;
    int b0 = m0 >> 11, tt0 = m0 & (TLEN - 1);
    int b1 = m1 >> 11, tt1 = m1 & (TLEN - 1);
    float u0 = acc0[e] + bc1;
    float u1 = acc1[e] + bc1;
    o1Pad[((size_t)b0 * SROWS + 1 + tt0) * 2048 + c] = (_Float16)(u0 > 0.0f ? u0 : 0.0f);
    o1Pad[((size_t)b1 * SROWS + 1 + tt1) * 2048 + c] = (_Float16)(u1 > 0.0f ? u1 : 0.0f);
  }
}

// head2: out2 = relu(conv_k3(out1, W2) + b2)  2048 -> 256, two M tiles per wave
__global__ void __launch_bounds__(256)
k_head2(const _Float16* __restrict__ o1Pad, const _Float16* __restrict__ W2P,
        const float* __restrict__ bias2, float* __restrict__ o2F32) {
  const int lane = threadIdx.x & 31;
  const int wave = threadIdx.x >> 5;
  const int hf = lane >> 4, ln = lane & 15;
  const int nTilesN = 256 / 16;                       // 16
  int tile = blockIdx.x * 8 + wave;                   // over (MT/32)*16
  int tileM = tile / nTilesN, tileN = tile % nTilesN;
  int mBase = tileM * 32, nBase = tileN * 16;
  int row0 = mBase + ln, row1 = row0 + 16;
  int b = row0 >> 11;
  int t0 = row0 & (TLEN - 1), t1 = row1 & (TLEN - 1);

  v8f acc0 = {}; v8f acc1 = {};
  for (int j = 0; j < 3; ++j) {
    const _Float16* aRow0 = o1Pad + ((size_t)b * SROWS + t0 + j) * 2048;
    const _Float16* aRow1 = o1Pad + ((size_t)b * SROWS + t1 + j) * 2048;
    const _Float16* bp = W2P + ((size_t)(j * 16 + tileN) * 64) * 512 + lane * 16;
#pragma unroll 8
    for (int kt = 0; kt < 64; ++kt) {
      Frag16 a0, a1, bf;
      load_a_frag(a0, aRow0 + kt * 32, hf);
      load_a_frag(a1, aRow1 + kt * 32, hf);
      load_b_packed(bf, bp + (size_t)kt * 512);
      WMMA_F16(acc0, a0, bf);
      WMMA_F16(acc1, a1, bf);
    }
  }
  int c = nBase + ln;
  float bc2 = bias2[c];
#pragma unroll
  for (int e = 0; e < 8; ++e) {
    int m0 = mBase + hf * 8 + e;
    int m1 = m0 + 16;
    float u0 = acc0[e] + bc2;
    float u1 = acc1[e] + bc2;
    o2F32[(size_t)m0 * 256 + c] = (u0 > 0.0f ? u0 : 0.0f);
    o2F32[(size_t)m1 * 256 + c] = (u1 > 0.0f ? u1 : 0.0f);
  }
}

// head3: out[m] = tanh(dot(out2[m,:], W3) + b3), one wave per row
__global__ void k_head3(const float* __restrict__ o2F32, const float* __restrict__ W3,
                        const float* __restrict__ b3, float* __restrict__ out) {
  int wave = threadIdx.x >> 5, lane = threadIdx.x & 31;
  int m = blockIdx.x * 8 + wave;
  const float4* p = (const float4*)(o2F32 + (size_t)m * 256);
  const float4* w = (const float4*)W3;
  float s = 0.0f;
#pragma unroll
  for (int i = 0; i < 2; ++i) {
    float4 a = p[lane * 2 + i];
    float4 ww = w[lane * 2 + i];
    s += a.x * ww.x + a.y * ww.y + a.z * ww.z + a.w * ww.w;
  }
#pragma unroll
  for (int off = 16; off > 0; off >>= 1) s += __shfl_xor(s, off, 32);
  if (lane == 0) out[m] = tanhf(s + b3[0]);
}

// ---------------- host launcher ----------------

extern "C" void kernel_launch(void* const* d_in, const int* in_sizes, int n_in,
                              void* d_out, int out_size, void* d_ws, size_t ws_size,
                              hipStream_t stream) {
  (void)in_sizes; (void)n_in; (void)out_size; (void)ws_size;
  const float* x     = (const float*)d_in[0];
  const float* cond  = (const float*)d_in[1];
  const float* Wc    = (const float*)d_in[2];
  const float* bc    = (const float*)d_in[3];
  const float* Wt    = (const float*)d_in[4];
  const float* bt    = (const float*)d_in[5];
  const float* Ws    = (const float*)d_in[6];
  const float* bs    = (const float*)d_in[7];
  const float* Dt    = (const float*)d_in[8];
  const float* Bt    = (const float*)d_in[9];
  const float* Ds    = (const float*)d_in[10];
  const float* Bs    = (const float*)d_in[11];
  const float* Wskip = (const float*)d_in[12];
  const float* bskip = (const float*)d_in[13];
  const float* Wres  = (const float*)d_in[14];
  const float* bres  = (const float*)d_in[15];
  const float* W1    = (const float*)d_in[16];
  const float* b1    = (const float*)d_in[17];
  const float* W2    = (const float*)d_in[18];
  const float* b2    = (const float*)d_in[19];
  const float* W3    = (const float*)d_in[20];
  const float* b3    = (const float*)d_in[21];
  float* out = (float*)d_out;

  char* ws = (char*)d_ws;
  size_t off = 0;
  auto carve = [&](size_t bytes) -> void* {
    off = (off + 255) & ~(size_t)255;
    void* p = ws + off;
    off += bytes;
    return p;
  };
  const size_t NE = (size_t)MT * CCH;
  const int nHPad = BB * HROWS * CCH;        // 3,145,728
  const int nSPad = BB * SROWS * CCH;        // 2,099,200
  float*    hF32   = (float*)   carve(NE * 4);
  _Float16* hPad   = (_Float16*)carve((size_t)nHPad * 2);
  _Float16* gF16   = (_Float16*)carve(NE * 2);
  float*    tt     = (float*)   carve(NE * 4);
  float*    ts     = (float*)   carve(NE * 4);
  float*    skipF  = (float*)   carve(NE * 4);
  _Float16* skPad  = (_Float16*)carve((size_t)nSPad * 2);
  _Float16* o1Pad  = (_Float16*)carve((size_t)BB * SROWS * 2048 * 2);
  float*    o2F32  = (float*)   carve((size_t)MT * 256 * 4);
  const int nWt  = LNUM * 3 * CCH * CCH;
  const int nW11 = LNUM * CCH * CCH;
  const int nW1  = 3 * CCH * 2048;
  const int nW2  = 3 * 2048 * 256;
  _Float16* WtP  = (_Float16*)carve((size_t)nWt * 2);
  _Float16* WsP  = (_Float16*)carve((size_t)nWt * 2);
  _Float16* WskP = (_Float16*)carve((size_t)nW11 * 2);
  _Float16* WreP = (_Float16*)carve((size_t)nW11 * 2);
  _Float16* W1P  = (_Float16*)carve((size_t)nW1 * 2);
  _Float16* W2P  = (_Float16*)carve((size_t)nW2 * 2);

  // pack weights into WMMA-B fragment-major f16
  auto packGrid = [](int nElem) { return (nElem / 16 + 255) / 256; };
  k_pack_b<<<packGrid(nWt), 256, 0, stream>>>(Wt, WtP, CCH, CCH, nWt / 16);
  k_pack_b<<<packGrid(nWt), 256, 0, stream>>>(Ws, WsP, CCH, CCH, nWt / 16);
  k_pack_b<<<packGrid(nW11), 256, 0, stream>>>(Wskip, WskP, CCH, CCH, nW11 / 16);
  k_pack_b<<<packGrid(nW11), 256, 0, stream>>>(Wres, WreP, CCH, CCH, nW11 / 16);
  k_pack_b<<<packGrid(nW1), 256, 0, stream>>>(W1, W1P, CCH, 2048, nW1 / 16);
  k_pack_b<<<packGrid(nW2), 256, 0, stream>>>(W2, W2P, 2048, 256, nW2 / 16);

  // zero pads, then fill interiors
  k_zero_f16<<<(nHPad + 255) / 256, 256, 0, stream>>>(hPad, nHPad);
  k_zero_f16<<<(nSPad + 255) / 256, 256, 0, stream>>>(skPad, nSPad);
  k_zero_o1_pads<<<(BB * 2 * 2048) / 256, 256, 0, stream>>>(o1Pad);
  k_input_proj<<<(int)(NE / 256), 256, 0, stream>>>(x, Wc, bc, hF32, hPad, skipF);

  const int gemmBlocks = (MT / 32) * (CCH / 16) / 8;    // 512
  for (int l = 0; l < LNUM; ++l) {
    int dil = 1 << l;
    k_cond_dense<<<(int)(NE / 4 / 256), 256, 0, stream>>>(cond, Dt, Bt, Ds, Bs, tt, ts, l);
    k_gated_conv<<<gemmBlocks, 256, 0, stream>>>(hPad, WtP, WsP, bt, bs, tt, ts, gF16, l, dil);
    k_skip_res<<<gemmBlocks, 256, 0, stream>>>(gF16, WskP, WreP, bskip, bres, skipF, hF32, hPad, l);
  }

  k_skip_to_pad<<<(int)(NE / 256), 256, 0, stream>>>(skipF, skPad);
  k_head1<<<(MT / 32) * (2048 / 16) / 8, 256, 0, stream>>>(skPad, W1P, b1, o1Pad);
  k_head2<<<(MT / 32) * (256 / 16) / 8, 256, 0, stream>>>(o1Pad, W2P, b2, o2F32);
  k_head3<<<MT / 8, 256, 0, stream>>>(o2F32, W3, b3, out);
}
